// GraphSAGE_81767587381364
// MI455X (gfx1250) — compile-verified, matrix operand order checked by reading, not run on hardware
//
#include <hip/hip_runtime.h>
#include <math.h>

typedef __attribute__((ext_vector_type(16))) __bf16 v16bf;
typedef __attribute__((ext_vector_type(8)))  float  v8f;

#define N_NODES 50000
#define N_EDGES 800000
#define D_IN    128
#define D_HID   128
#define D_OUT   64

// ---------- helpers ----------

__device__ __forceinline__ unsigned short f2bf(float f) {
    // round-to-nearest-even f32 -> bf16 (finite inputs)
    unsigned int u = __float_as_uint(f);
    u += 0x7FFFu + ((u >> 16) & 1u);
    return (unsigned short)(u >> 16);
}

__device__ __forceinline__ float bf2f(unsigned short s) {
    return __uint_as_float(((unsigned int)s) << 16);
}

union Frag {
    v16bf v;
    uint4 q[2];
};

// ---------- weight pre-pack: f32 KxN (row-major) -> bf16 B-fragment order ----------
// Packed index = ((ct*ksteps + ks)*32 + lane)*16 + j
//   col n = ct*16 + (lane&15)
//   k     = ks*32 + (lane>>4)*8 + (j<8 ? j : j+8)
__global__ void pack_w_kernel(const float* __restrict__ W, unsigned short* __restrict__ Wp,
                              int K, int N) {
    int total = (N >> 4) * (K >> 5) * 512;
    int idx = blockIdx.x * blockDim.x + threadIdx.x;
    if (idx >= total) return;
    int j      = idx & 15;
    int lane   = (idx >> 4) & 31;
    int t      = idx >> 9;
    int ksteps = K >> 5;
    int ks = t % ksteps;
    int ct = t / ksteps;
    int n = ct * 16 + (lane & 15);
    int k = ks * 32 + ((lane >> 4) << 3) + ((j < 8) ? j : j + 8);
    Wp[idx] = f2bf(W[k * N + n]);
}

// ---------- elementwise ----------
__global__ void zero_kernel(float* __restrict__ p, int n) {
    int i = blockIdx.x * blockDim.x + threadIdx.x;
    if (i < n) p[i] = 0.0f;
}

__global__ void cvt_bf16_kernel(const float* __restrict__ src,
                                unsigned short* __restrict__ dst, int n) {
    int i = blockIdx.x * blockDim.x + threadIdx.x;
    if (i < n) dst[i] = f2bf(src[i]);
}

// meanbf[i] = bf16( agg[i] / max(cnt[row],1) ), D = 128 fixed
__global__ void mean_bf16_kernel(const float* __restrict__ agg,
                                 const float* __restrict__ cnt,
                                 unsigned short* __restrict__ meanbf, int n) {
    int i = blockIdx.x * blockDim.x + threadIdx.x;
    if (i >= n) return;
    float c = cnt[i >> 7];
    meanbf[i] = f2bf(agg[i] * (1.0f / fmaxf(c, 1.0f)));
}

// ---------- edge scatter: one wave per edge, bf16 gather (256B/row), f32 L2 atomics ----------
__global__ void scatter_kernel(const unsigned short* __restrict__ xbf,
                               const long long* __restrict__ src,
                               const long long* __restrict__ dst,
                               float* __restrict__ agg,
                               float* __restrict__ cnt,
                               int nEdges, int doCount) {
    int wave = (blockIdx.x * blockDim.x + threadIdx.x) >> 5;
    int lane = threadIdx.x & 31;
    if (wave >= nEdges) return;
    long long s = src[wave];
    long long d = dst[wave];
    ushort4 v = *(const ushort4*)(xbf + (size_t)s * D_HID + lane * 4);
    float* a = agg + (size_t)d * D_HID + lane * 4;
    atomicAdd(a + 0, bf2f(v.x));
    atomicAdd(a + 1, bf2f(v.y));
    atomicAdd(a + 2, bf2f(v.z));
    atomicAdd(a + 3, bf2f(v.w));
    if (doCount && lane == 0) atomicAdd(cnt + d, 1.0f);
}

// ---------- fused SAGE layer: out = act( mean@Wl + x@Wr + b ), pure-WMMA inner loop ----------
// block = 32 * (Dout/16); wave ct owns a 16x16 tile; grid.x = N/16. All bf16 pre-converted.
template <int HAS_AGG, int ACT, int OUT_BF16, int KSTEPS>  // ACT: 0=ReLU, 1=sigmoid
__global__ void sage_wmma_kernel(const unsigned short* __restrict__ xbf,
                                 const unsigned short* __restrict__ meanbf,
                                 const unsigned short* __restrict__ Wlp,
                                 const unsigned short* __restrict__ Wrp,
                                 const float* __restrict__ bias,
                                 void* __restrict__ outv,
                                 int Dout) {
    const int lane     = threadIdx.x & 31;
    const int ct       = threadIdx.x >> 5;   // column tile
    const int nodeBase = blockIdx.x << 4;
    const int row      = nodeBase + (lane & 15);
    const int kgrp     = (lane >> 4) << 3;   // 0 or 8
    const size_t xrow  = (size_t)row * (KSTEPS * 32);
    const size_t wbase = (size_t)ct * KSTEPS;

    v8f acc = {};
#pragma unroll
    for (int ks = 0; ks < KSTEPS; ++ks) {
        const int kb = ks * 32 + kgrp;
        // self term: x @ Wr
        {
            Frag a, b;
            a.q[0] = *(const uint4*)(xbf + xrow + kb);
            a.q[1] = *(const uint4*)(xbf + xrow + kb + 16);
            const uint4* w = (const uint4*)(Wrp + ((wbase + ks) * 32 + lane) * 16);
            b.q[0] = w[0];
            b.q[1] = w[1];
            acc = __builtin_amdgcn_wmma_f32_16x16x32_bf16(
                false, a.v, false, b.v, (short)0, acc, false, false);
        }
        // neighbor-mean term: mean @ Wl
        if (HAS_AGG) {
            Frag m, b;
            m.q[0] = *(const uint4*)(meanbf + xrow + kb);
            m.q[1] = *(const uint4*)(meanbf + xrow + kb + 16);
            const uint4* w = (const uint4*)(Wlp + ((wbase + ks) * 32 + lane) * 16);
            b.q[0] = w[0];
            b.q[1] = w[1];
            acc = __builtin_amdgcn_wmma_f32_16x16x32_bf16(
                false, m.v, false, b.v, (short)0, acc, false, false);
        }
    }

    // C/D layout: lanes 0-15 -> N=lane, VGPR r -> M=r; lanes 16-31 -> N=lane-16, M=r+8
    const int col  = ct * 16 + (lane & 15);
    const int mOff = (lane >> 4) << 3;
    const float bv = bias[col];
#pragma unroll
    for (int r = 0; r < 8; ++r) {
        float v = acc[r] + bv;
        if (ACT == 0) {
            v = fmaxf(v, 0.0f);
        } else {
            v = 1.0f / (1.0f + expf(-v));
        }
        size_t idx = (size_t)(nodeBase + r + mOff) * Dout + col;
        if (OUT_BF16) {
            ((unsigned short*)outv)[idx] = f2bf(v);
        } else {
            ((float*)outv)[idx] = v;
        }
    }
}

// ---------- launch ----------
extern "C" void kernel_launch(void* const* d_in, const int* in_sizes, int n_in,
                              void* d_out, int out_size, void* d_ws, size_t ws_size,
                              hipStream_t stream) {
    const float*     x    = (const float*)d_in[0];
    const long long* ei   = (const long long*)d_in[1];  // int64 edge_index [2, E]
    const float*     Wl1  = (const float*)d_in[2];
    const float*     b1   = (const float*)d_in[3];
    const float*     Wr1  = (const float*)d_in[4];
    const float*     Wl2  = (const float*)d_in[5];
    const float*     b2   = (const float*)d_in[6];
    const float*     Wr2  = (const float*)d_in[7];
    const float*     Wout = (const float*)d_in[8];
    const float*     bout = (const float*)d_in[9];
    float*           out  = (float*)d_out;

    const long long* src = ei;
    const long long* dst = ei + N_EDGES;

    const int feat = N_NODES * D_HID;  // 6.4M elements

    // workspace layout (f32 first, then bf16; every section 16B/32B aligned)
    float* agg = (float*)d_ws;                        // N*128 f32
    float* cnt = agg + (size_t)feat;                  // N    f32
    unsigned short* xbf    = (unsigned short*)(cnt + N_NODES);  // N*128 bf16
    unsigned short* meanbf = xbf + (size_t)feat;                // N*128 bf16 (reused both layers)
    unsigned short* h1bf   = meanbf + (size_t)feat;             // N*128 bf16
    unsigned short* h2bf   = h1bf + (size_t)feat;               // N*128 bf16
    unsigned short* wl1p   = h2bf + (size_t)feat;
    unsigned short* wr1p   = wl1p + 8 * 4 * 512;      // (128/16)*(128/32)*512
    unsigned short* wl2p   = wr1p + 8 * 4 * 512;
    unsigned short* wr2p   = wl2p + 8 * 4 * 512;
    unsigned short* woutp  = wr2p + 8 * 4 * 512;      // (64/16)*(128/32)*512

    // 1) pack weights to bf16 fragment order
    const int totHid = (D_HID / 16) * (D_IN / 32) * 512;   // 16384
    const int totOut = (D_OUT / 16) * (D_HID / 32) * 512;  // 8192
    pack_w_kernel<<<(totHid + 255) / 256, 256, 0, stream>>>(Wl1, wl1p, D_IN, D_HID);
    pack_w_kernel<<<(totHid + 255) / 256, 256, 0, stream>>>(Wr1, wr1p, D_IN, D_HID);
    pack_w_kernel<<<(totHid + 255) / 256, 256, 0, stream>>>(Wl2, wl2p, D_HID, D_HID);
    pack_w_kernel<<<(totHid + 255) / 256, 256, 0, stream>>>(Wr2, wr2p, D_HID, D_HID);
    pack_w_kernel<<<(totOut + 255) / 256, 256, 0, stream>>>(Wout, woutp, D_HID, D_OUT);

    const int eb = (N_EDGES + 7) / 8;   // 8 edge-waves per 256-thread block
    const int nodeTiles = N_NODES / 16; // 3125 exact
    const int g = (feat + 255) / 256;

    // 2) x -> bf16 once
    cvt_bf16_kernel<<<g, 256, 0, stream>>>(x, xbf, feat);

    // 3) layer 1: zero agg+cnt, scatter bf16 rows with f32 atomics, mean->bf16, WMMA layer
    zero_kernel<<<(feat + N_NODES + 255) / 256, 256, 0, stream>>>(agg, feat + N_NODES);
    scatter_kernel<<<eb, 256, 0, stream>>>(xbf, src, dst, agg, cnt, N_EDGES, 1);
    mean_bf16_kernel<<<g, 256, 0, stream>>>(agg, cnt, meanbf, feat);
    sage_wmma_kernel<1, 0, 1, 4><<<nodeTiles, 32 * (D_HID / 16), 0, stream>>>(
        xbf, meanbf, wl1p, wr1p, b1, h1bf, D_HID);

    // 4) layer 2
    zero_kernel<<<g, 256, 0, stream>>>(agg, feat);
    scatter_kernel<<<eb, 256, 0, stream>>>(h1bf, src, dst, agg, cnt, N_EDGES, 0);
    mean_bf16_kernel<<<g, 256, 0, stream>>>(agg, cnt, meanbf, feat);
    sage_wmma_kernel<1, 0, 1, 4><<<nodeTiles, 32 * (D_HID / 16), 0, stream>>>(
        h1bf, meanbf, wl2p, wr2p, b2, h2bf, D_HID);

    // 5) output head: sigmoid(h2 @ Wout + bout) -> f32 d_out
    sage_wmma_kernel<0, 1, 0, 4><<<nodeTiles, 32 * (D_OUT / 16), 0, stream>>>(
        h2bf, (const unsigned short*)nullptr,
        (const unsigned short*)nullptr, woutp, bout, out, D_OUT);
}
